// Encoder_64665027609415
// MI455X (gfx1250) — matrix operand order repaired and checked by reading, not compiled, not run on hardware
//
#include <hip/hip_runtime.h>
#include <hip/hip_bf16.h>
#include <stdint.h>

// ---------------------------------------------------------------------------
// LSTM encoder for MI455X (gfx1250, wave32, WMMA).
// Phase 1: xz = x @ W_in + b_lstm            (bf16 WMMA, M=32768 K=1024 N=2048)
// Phase 2: 256-step LSTM recurrence, 128 WGs (8 batch groups x 16 U-slices),
//          h exchanged via per-t global buffer + per-group atomic barrier.
// Phase 3: softmax(relu((h*mask) @ W_dense + b)) with mask folded into the
//          A-fragment load (per-lane scalar scale).
// All GEMMs: v_wmma_f32_16x16x32_bf16, fp32 accumulate.
// ---------------------------------------------------------------------------

typedef __attribute__((ext_vector_type(16))) __bf16 bf16x16;
typedef __attribute__((ext_vector_type(8)))  float  f32x8;

union FragBF { uint32_t u[8]; bf16x16 v; };

__device__ __forceinline__ uint32_t pk_bf16(float a, float b) {
  uint32_t ua = __float_as_uint(a);
  uint32_t ub = __float_as_uint(b);
  ua = (ua + 0x7FFFu + ((ua >> 16) & 1u)) >> 16;   // RNE round to bf16
  ub = (ub + 0x7FFFu + ((ub >> 16) & 1u)) >> 16;
  return ua | (ub << 16);
}
__device__ __forceinline__ uint16_t f2bf(float a) {
  uint32_t ua = __float_as_uint(a);
  return (uint16_t)((ua + 0x7FFFu + ((ua >> 16) & 1u)) >> 16);
}
__device__ __forceinline__ float bf2f(uint16_t h) {
  return __uint_as_float(((uint32_t)h) << 16);
}
// A-fragment (16x32 bf16) K of dword j, per ISA 7.12.2 16-bit A layout.
__device__ __forceinline__ int amap(int j, int half) {
  return ((j & 4) ? 16 : 0) + half * 8 + (j & 3) * 2;
}
// B-fragment (32x16 bf16) K of dword j: same K order as the A fragment (amap).
__device__ __forceinline__ int bmap(int j, int half) {
  return ((j & 4) ? 16 : 0) + half * 8 + (j & 3) * 2;
}
__device__ __forceinline__ float sigmoid_fast(float x) {
  return 1.0f / (1.0f + __expf(-x));
}
__device__ __forceinline__ float tanh_fast(float x) {
  return 2.0f / (1.0f + __expf(-2.0f * x)) - 1.0f;
}

// ---------------------------------------------------------------------------
// Pack an fp32 [K,N] weight matrix into bf16 B-fragment order:
// dst[((kb*(N/16)+nt)*32 + lane)*8 + j]  -> two global_load_b128 per fragment.
// ---------------------------------------------------------------------------
__global__ void __launch_bounds__(256) pack_b_kernel(
    const float* __restrict__ W, uint32_t* __restrict__ dst, int K, int N) {
  int tid = blockIdx.x * blockDim.x + threadIdx.x;
  int total = (K / 32) * (N / 16) * 32 * 8;
  if (tid >= total) return;
  int j    = tid & 7;
  int lane = (tid >> 3) & 31;
  int nt   = (tid >> 8) % (N / 16);
  int kb   = tid / (256 * (N / 16));
  int n    = nt * 16 + (lane & 15);
  int half = lane >> 4;
  int k0   = kb * 32 + bmap(j, half);
  float a = W[(size_t)k0 * N + n];
  float b = W[(size_t)(k0 + 1) * N + n];
  dst[tid] = pk_bf16(a, b);
}

__global__ void init_cnt_kernel(int* cnt) {
  if (threadIdx.x < 8) cnt[threadIdx.x] = 0;
}

// ---------------------------------------------------------------------------
// Phase 1: xz[T*B, 2048] = x[T*B, 1024] @ W_in + b_lstm   (stored bf16)
// Block tile 64x256, 8 waves as 2x4, each wave 2x4 16x16 tiles.
// x staged fp32->bf16 through LDS (padded stride 34 -> conflict-free frags).
// ---------------------------------------------------------------------------
#define XSTR 34
__global__ void __launch_bounds__(256) xz_gemm_kernel(
    const float* __restrict__ x, const uint32_t* __restrict__ WinPack,
    const float* __restrict__ b_lstm, uint16_t* __restrict__ xz) {
  __shared__ uint16_t xs[64 * XSTR];
  const int n0   = blockIdx.x * 256;
  const int m0   = blockIdx.y * 64;
  const int tid  = threadIdx.x;
  const int wave = tid >> 5, lane = tid & 31;
  const int ml   = lane & 15, half = lane >> 4;
  const int mw   = wave >> 2, nw = wave & 3;

  f32x8 acc[2][4];
  for (int a = 0; a < 2; a++)
    for (int b = 0; b < 4; b++)
      for (int j = 0; j < 8; j++) acc[a][b][j] = 0.0f;

  for (int kb = 0; kb < 32; ++kb) {
    { // cooperative stage: 64 rows x 32 K fp32 -> bf16 LDS
      int row = tid >> 2;
      int kq  = (tid & 3) * 8;
      const float4* src =
          (const float4*)(x + (size_t)(m0 + row) * 1024 + kb * 32 + kq);
      float4 v0 = src[0], v1 = src[1];
      uint32_t* d = (uint32_t*)&xs[row * XSTR + kq];
      d[0] = pk_bf16(v0.x, v0.y);
      d[1] = pk_bf16(v0.z, v0.w);
      d[2] = pk_bf16(v1.x, v1.y);
      d[3] = pk_bf16(v1.z, v1.w);
    }
    __syncthreads();
    FragBF afr[2];
#pragma unroll
    for (int mt = 0; mt < 2; ++mt) {
      int row = mw * 32 + mt * 16 + ml;
#pragma unroll
      for (int j = 0; j < 8; j++)
        afr[mt].u[j] = *(const uint32_t*)&xs[row * XSTR + amap(j, half)];
    }
#pragma unroll
    for (int nt = 0; nt < 4; ++nt) {
      int ntg = (n0 >> 4) + nw * 4 + nt;
      const uint4* bp =
          (const uint4*)(WinPack + ((size_t)(kb * 128 + ntg) * 32 + lane) * 8);
      FragBF bfr;
      *(uint4*)&bfr.u[0] = bp[0];
      *(uint4*)&bfr.u[4] = bp[1];
#pragma unroll
      for (int mt = 0; mt < 2; ++mt)
        acc[mt][nt] = __builtin_amdgcn_wmma_f32_16x16x32_bf16(
            false, afr[mt].v, false, bfr.v, (short)0, acc[mt][nt], false, false);
    }
    __syncthreads();
  }
#pragma unroll
  for (int mt = 0; mt < 2; ++mt)
#pragma unroll
    for (int nt = 0; nt < 4; ++nt) {
      int col = n0 + nw * 64 + nt * 16 + ml;
      float bias = b_lstm[col];
#pragma unroll
      for (int j = 0; j < 8; j++) {
        int row = m0 + mw * 32 + mt * 16 + j + half * 8;
        xz[(size_t)row * 2048 + col] = f2bf(acc[mt][nt][j] + bias);
      }
    }
}

// ---------------------------------------------------------------------------
// Phase 2: recurrence on 128 WGs = 8 batch groups x 16 U-slices (32 cols).
// Wave w owns one 16x16 z-tile: gate = w/2, sub-slice = w%2 (16 cols).
// h(t) written to per-t global buffer hs (bf16); per-group monotonic atomic
// barrier + threadfence orders producers/consumers. Gates reassembled per
// (row,u) through an 8KB LDS z-buffer; c lives in the owning thread's regs.
// ---------------------------------------------------------------------------
#define NSLICE 16
__global__ void __launch_bounds__(256) lstm_seq_kernel(
    const uint16_t* __restrict__ xz, const uint32_t* __restrict__ WrecPack,
    const float* __restrict__ mask, uint16_t* __restrict__ hs,
    int* __restrict__ cnt) {
  __shared__ float zbuf[4 * 16 * 32];  // [gate][row][u]
  const int wg   = blockIdx.x;   // 0..127
  const int grp  = wg >> 4;      // batch group (16 rows)
  const int sl   = wg & 15;      // U slice (32 cols)
  const int r0   = grp * 16;
  const int us0  = sl * 32;
  const int tid  = threadIdx.x;
  const int wave = tid >> 5, lane = tid & 31;
  const int ml   = lane & 15, half = lane >> 4;
  const int gate = wave >> 1, sub = wave & 1;
  const int ntp  = gate * 32 + sl * 2 + sub;  // packed n-tile = (gate*512+us0+sub*16)/16
  // elementwise ownership: 2 consecutive cols per thread
  const int erow = tid >> 4;
  const int ecol = (tid & 15) * 2;
  float creg[2] = {0.0f, 0.0f};

  for (int t = 0; t < 256; ++t) {
    f32x8 acc;
    for (int j = 0; j < 8; j++) acc[j] = 0.0f;
    if (t > 0) {
      const uint16_t* hprev = hs + (size_t)(t - 1) * 128 * 512;
      for (int kb = 0; kb < 16; ++kb) {
        FragBF afr;
#pragma unroll
        for (int j = 0; j < 8; j++)
          afr.u[j] = *(const uint32_t*)(hprev + (size_t)(r0 + ml) * 512 +
                                        kb * 32 + amap(j, half));
        const uint4* bp = (const uint4*)(WrecPack +
                            ((size_t)(kb * 128 + ntp) * 32 + lane) * 8);
        FragBF bfr;
        *(uint4*)&bfr.u[0] = bp[0];
        *(uint4*)&bfr.u[4] = bp[1];
        acc = __builtin_amdgcn_wmma_f32_16x16x32_bf16(
            false, afr.v, false, bfr.v, (short)0, acc, false, false);
      }
    }
    // scatter z tile into LDS for gate reassembly
#pragma unroll
    for (int j = 0; j < 8; j++) {
      int m = j + half * 8;
      zbuf[(gate * 16 + m) * 32 + sub * 16 + ml] = acc[j];
    }
    __syncthreads();

    // elementwise LSTM cell for 2 (row,u) elements per thread
#pragma unroll
    for (int q = 0; q < 2; ++q) {
      int u = ecol + q;
      size_t zrow = ((size_t)t * 128 + r0 + erow) * 2048 + us0 + u;
      float zi = zbuf[(0 * 16 + erow) * 32 + u] + bf2f(xz[zrow + 0 * 512]);
      float zf = zbuf[(1 * 16 + erow) * 32 + u] + bf2f(xz[zrow + 1 * 512]);
      float zg = zbuf[(2 * 16 + erow) * 32 + u] + bf2f(xz[zrow + 2 * 512]);
      float zo = zbuf[(3 * 16 + erow) * 32 + u] + bf2f(xz[zrow + 3 * 512]);
      float ig = sigmoid_fast(zi);
      float fg = sigmoid_fast(zf);
      float gg = tanh_fast(zg);
      float og = sigmoid_fast(zo);
      float cn = fg * creg[q] + ig * gg;
      creg[q]  = cn;
      float h  = og * tanh_fast(cn);
      hs[((size_t)t * 128 + r0 + erow) * 512 + us0 + u] = f2bf(h);
    }

    // ---- per-group device barrier: publish h(t), arrive, spin ----
    __threadfence();   // make hs stores device-visible
    __syncthreads();   // whole WG done (also protects zbuf reuse)
    if (tid == 0) {
      __hip_atomic_fetch_add(&cnt[grp], 1, __ATOMIC_RELEASE,
                             __HIP_MEMORY_SCOPE_AGENT);
      int target = NSLICE * (t + 1);
      while (__hip_atomic_load(&cnt[grp], __ATOMIC_ACQUIRE,
                               __HIP_MEMORY_SCOPE_AGENT) < target)
        __builtin_amdgcn_s_sleep(1);
    }
    __syncthreads();
  }
}

// ---------------------------------------------------------------------------
// Phase 3: out = softmax(relu((h*mask) @ W_dense + b_dense)).
// WG = 16 rows x full N=1024; mask folded into A-fragment (per-lane scalar).
// ---------------------------------------------------------------------------
__global__ void __launch_bounds__(256) dense_softmax_kernel(
    const uint16_t* __restrict__ hs, const float* __restrict__ mask,
    const uint32_t* __restrict__ WdPack, const float* __restrict__ b_dense,
    float* __restrict__ out) {
  __shared__ float logits[16 * 1024];
  const int r0   = blockIdx.x * 16;
  const int tid  = threadIdx.x;
  const int wave = tid >> 5, lane = tid & 31;
  const int ml   = lane & 15, half = lane >> 4;
  const float mk = mask[r0 + ml];  // per-lane row mask

  f32x8 acc[8];
#pragma unroll
  for (int nt = 0; nt < 8; ++nt) {
    float bias = b_dense[wave * 128 + nt * 16 + ml];
    for (int j = 0; j < 8; j++) acc[nt][j] = bias;
  }
  for (int kb = 0; kb < 16; ++kb) {
    FragBF afr;
#pragma unroll
    for (int j = 0; j < 8; j++) {
      uint32_t d = *(const uint32_t*)(hs + (size_t)(r0 + ml) * 512 + kb * 32 +
                                      amap(j, half));
      float lo = bf2f((uint16_t)(d & 0xFFFFu)) * mk;
      float hi = bf2f((uint16_t)(d >> 16)) * mk;
      afr.u[j] = pk_bf16(lo, hi);
    }
#pragma unroll
    for (int nt = 0; nt < 8; ++nt) {
      int ntg = wave * 8 + nt;
      const uint4* bp =
          (const uint4*)(WdPack + ((size_t)(kb * 64 + ntg) * 32 + lane) * 8);
      FragBF bfr;
      *(uint4*)&bfr.u[0] = bp[0];
      *(uint4*)&bfr.u[4] = bp[1];
      acc[nt] = __builtin_amdgcn_wmma_f32_16x16x32_bf16(
          false, afr.v, false, bfr.v, (short)0, acc[nt], false, false);
    }
  }
#pragma unroll
  for (int nt = 0; nt < 8; ++nt) {
    int n = wave * 128 + nt * 16 + ml;
#pragma unroll
    for (int j = 0; j < 8; j++) {
      int m = j + half * 8;
      logits[m * 1024 + n] = fmaxf(acc[nt][j], 0.0f);  // relu
    }
  }
  __syncthreads();

  for (int rr = 0; rr < 2; ++rr) {  // each wave: 2 rows, lane-strided softmax
    int m = wave * 2 + rr;
    float* row = &logits[m * 1024];
    float mx = -1e30f;
    for (int n2 = lane; n2 < 1024; n2 += 32) mx = fmaxf(mx, row[n2]);
    for (int off = 16; off; off >>= 1) mx = fmaxf(mx, __shfl_xor(mx, off));
    float sum = 0.0f;
    for (int n2 = lane; n2 < 1024; n2 += 32) {
      float e = __expf(row[n2] - mx);
      sum += e;
      row[n2] = e;
    }
    for (int off = 16; off; off >>= 1) sum += __shfl_xor(sum, off);
    float inv = 1.0f / sum;
    float* o = out + (size_t)(r0 + m) * 1024;
    for (int n2 = lane; n2 < 1024; n2 += 32) o[n2] = row[n2] * inv;
  }
}

// ---------------------------------------------------------------------------
extern "C" void kernel_launch(void* const* d_in, const int* in_sizes, int n_in,
                              void* d_out, int out_size, void* d_ws,
                              size_t ws_size, hipStream_t stream) {
  (void)in_sizes; (void)n_in; (void)out_size; (void)ws_size;
  const float* x       = (const float*)d_in[0];  // [256,128,1024]
  const float* mask    = (const float*)d_in[1];  // [256,128]
  const float* W_in    = (const float*)d_in[2];  // [1024,2048]
  const float* W_rec   = (const float*)d_in[3];  // [512,2048]
  const float* b_lstm  = (const float*)d_in[4];  // [2048]
  const float* W_dense = (const float*)d_in[5];  // [512,1024]
  const float* b_dense = (const float*)d_in[6];  // [1024]
  float* out = (float*)d_out;

  // workspace: xz bf16 128MB | Wrec pack 2MB | Wdense pack 1MB | Win pack 4MB
  //          | h_seq bf16 32MB | barrier counters          (~167 MB total)
  char* ws = (char*)d_ws;
  uint16_t* xz  = (uint16_t*)(ws);
  uint32_t* wrp = (uint32_t*)(ws + 134217728);
  uint32_t* wdp = (uint32_t*)(ws + 134217728 + 2097152);
  uint32_t* wip = (uint32_t*)(ws + 134217728 + 2097152 + 1048576);
  uint16_t* hsq = (uint16_t*)(ws + 134217728 + 2097152 + 1048576 + 4194304);
  int*      cnt = (int*)(ws + 134217728 + 2097152 + 1048576 + 4194304 +
                         33554432);

  init_cnt_kernel<<<1, 32, 0, stream>>>(cnt);
  // pack weights into bf16 B-fragment layout (one-time per call, L2-resident)
  pack_b_kernel<<<4096, 256, 0, stream>>>(W_in,    wip, 1024, 2048);
  pack_b_kernel<<<2048, 256, 0, stream>>>(W_rec,   wrp,  512, 2048);
  pack_b_kernel<<<1024, 256, 0, stream>>>(W_dense, wdp,  512, 1024);

  // phase 1: input projection (fully parallel big GEMM)
  xz_gemm_kernel<<<dim3(8, 512), 256, 0, stream>>>(x, wip, b_lstm, xz);

  // phase 2: recurrence (128 WGs, per-group device barrier each step)
  lstm_seq_kernel<<<128, 256, 0, stream>>>(xz, wrp, mask, hsq, cnt);

  // phase 3: dense + fused softmax (mask applied on A-fragment load)
  dense_softmax_kernel<<<2048, 256, 0, stream>>>(hsq, mask, wdp, b_dense, out);
}
